// RNN_1597727834309
// MI455X (gfx1250) — compile-verified, matrix operand order checked by reading, not run on hardware
//
#include <hip/hip_runtime.h>
#include <hip/hip_bf16.h>

// ---------------------------------------------------------------------------
// 2-layer tanh Elman RNN for MI455X (gfx1250, wave32, WMMA).
//   Phase 0: one-shot f32->f16 conversion of input + all weights (vectorized).
//   Phase 1 (parallel, whole chip): Y0 = input @ W_ih_l0^T + b_ih_l0 + b_hh_l0
//       pure-f16 WMMA GEMM, 2Mx4N tiles/wave; Y0 stored TRANSPOSED (H x L*N)
//       so phase-2 accumulator init is one b128 load per tile per lane.
//   Phase 2 (serial, 1 persistent workgroup on 1 WGP, state in LDS):
//     h0[t] = tanh(Y0[t] + h0[t-1] @ W_hh_l0^T)
//     h1[t] = tanh(h0[t] @ W_ih_l1^T + h1[t-1] @ W_hh_l1^T + b1)
//       2Mx2N tiles per wave, k-loop outermost for A/B fragment reuse.
// All GEMMs use v_wmma_f32_16x16x32_f16 (f16 operands, f32 accumulate).
// ---------------------------------------------------------------------------

typedef __attribute__((ext_vector_type(16))) _Float16 v16h;
typedef __attribute__((ext_vector_type(8)))  _Float16 v8h;
typedef __attribute__((ext_vector_type(8)))  float    v8f;
typedef __attribute__((ext_vector_type(4)))  float    v4f;

#define RNN_L 512
#define RNN_N 64
#define RNN_H 512
#define RNN_I 512
#define RNN_LN (RNN_L * RNN_N)   // 32768

// Load a 16x32 f16 A/B fragment from a row-major f16 array (row stride in
// elements).  ISA layout (wave32): lane holds row (lane&15); two 16B chunks
// at K = kb..kb+7 and K = kb+16..kb+23, kb = (lane>>4)*8.
__device__ inline v16h load_frag_h16(const _Float16* __restrict__ base,
                                     int stride, int lane) {
  const int r  = lane & 15;
  const int kb = (lane >> 4) << 3;
  const _Float16* p = base + r * stride + kb;
  union { v16h v; struct { v8h lo; v8h hi; } s; } u;
  u.s.lo = *(const v8h*)p;          // K = kb .. kb+7    (16B)
  u.s.hi = *(const v8h*)(p + 16);   // K = kb+16 .. kb+23 (16B)
  return u.v;
}

__device__ inline v8f wmma_f16(v16h a, v16h b, v8f c) {
  return __builtin_amdgcn_wmma_f32_16x16x32_f16(
      /*neg_a=*/false, a, /*neg_b=*/false, b,
      /*c_mod=*/(short)0, c, /*reuse_a=*/false, /*reuse_b=*/false);
}

// ------------------- phase 0: vectorized f32 -> f16 ------------------------
// n8 = element count / 8; each thread converts 8 elements (32B -> 16B).
__global__ void f32_to_f16_vec8_kernel(const float* __restrict__ src,
                                       _Float16* __restrict__ dst, int n8) {
  const int i = blockIdx.x * blockDim.x + threadIdx.x;
  if (i >= n8) return;
  const v4f x0 = ((const v4f*)src)[2 * i];
  const v4f x1 = ((const v4f*)src)[2 * i + 1];
  v8h o;
#pragma unroll
  for (int j = 0; j < 4; ++j) {
    o[j]     = (_Float16)x0[j];
    o[4 + j] = (_Float16)x1[j];
  }
  ((v8h*)dst)[i] = o;
}

// ------------------- phase 1: big parallel input GEMM ----------------------
// Y0t[H x (L*N)] (f16, transposed) = (x @ Wih^T + b_ih + b_hh)^T.
// Each wave: 2 M-tiles x 4 N-tiles (32x64 output), k-loop outermost.
__global__ void __launch_bounds__(1024)
rnn_pre_kernel(const _Float16* __restrict__ xh,    // (L*N, I) f16
               const _Float16* __restrict__ Wih,   // (H, I)  f16
               const float* __restrict__ bih,
               const float* __restrict__ bhh,
               _Float16* __restrict__ Y0t) {       // (H, L*N) f16
  const int lane = threadIdx.x & 31;
  const int wave = threadIdx.x >> 5;
  const int grp  = blockIdx.x * 32 + wave;   // 8192 groups
  const int mg = grp >> 3;                   // 0..1023 -> rows mg*32
  const int ng = grp & 7;                    // 0..7    -> cols ng*64
  const int m0 = mg * 32;
  const int n0 = ng * 64;
  const int roff = (lane >> 4) << 3;
  const int lc   = lane & 15;

  v8f acc[2][4];
#pragma unroll
  for (int nt = 0; nt < 4; ++nt) {
    const int col = n0 + nt * 16 + lc;
    const float bsum = bih[col] + bhh[col];
#pragma unroll
    for (int mt = 0; mt < 2; ++mt)
#pragma unroll
      for (int i = 0; i < 8; ++i) acc[mt][nt][i] = bsum;
  }

  for (int k0 = 0; k0 < RNN_I; k0 += 32) {
    const v16h a0 = load_frag_h16(xh + (size_t)m0 * RNN_I + k0, RNN_I, lane);
    const v16h a1 = load_frag_h16(xh + (size_t)(m0 + 16) * RNN_I + k0, RNN_I, lane);
#pragma unroll
    for (int nt = 0; nt < 4; ++nt) {
      const v16h b = load_frag_h16(Wih + (size_t)(n0 + nt * 16) * RNN_I + k0,
                                   RNN_I, lane);
      acc[0][nt] = wmma_f16(a0, b, acc[0][nt]);
      acc[1][nt] = wmma_f16(a1, b, acc[1][nt]);
    }
  }

  // Transposed store: per lane, the 8 accumulator rows are contiguous -> one
  // 16B store per tile.
#pragma unroll
  for (int mt = 0; mt < 2; ++mt)
#pragma unroll
    for (int nt = 0; nt < 4; ++nt) {
      const int row0 = m0 + mt * 16 + roff;        // multiple of 8
      const int col  = n0 + nt * 16 + lc;
      v8h o;
#pragma unroll
      for (int i = 0; i < 8; ++i) o[i] = (_Float16)acc[mt][nt][i];
      *(v8h*)(Y0t + (size_t)col * RNN_LN + row0) = o;
    }
}

// ---------------- phase 2: persistent recurrent workgroup ------------------
// 1024 threads = 32 waves; wave = (mg in 0..1) x (ng in 0..15):
//   owns rows [32*mg, 32*mg+32) x cols [32*ng, 32*ng+32)  (2x2 16x16 tiles).
// LDS: h0 ping-pong (2 x 64KB) + h1 (64KB) = 192KB dynamic.
__global__ void __launch_bounds__(1024)
rnn_recur_kernel(const _Float16* __restrict__ Y0t,      // (H, L*N) f16 preact
                 const _Float16* __restrict__ Whh0,     // (H,H) f16
                 const _Float16* __restrict__ Wih1,     // (H,H) f16
                 const _Float16* __restrict__ Whh1,     // (H,H) f16
                 const float* __restrict__ bih1,
                 const float* __restrict__ bhh1,
                 const float* __restrict__ h_init,      // (2,N,H) f32
                 float* __restrict__ out,               // (L,N,H) f32
                 float* __restrict__ hn) {              // (2,N,H) f32
  extern __shared__ _Float16 smem[];
  _Float16* h0buf0 = smem;                     // N*H
  _Float16* h0buf1 = smem + RNN_N * RNN_H;     // N*H
  _Float16* h1buf  = smem + 2 * RNN_N * RNN_H; // N*H

  const int tid  = threadIdx.x;
  const int lane = tid & 31;
  const int wave = tid >> 5;

  // Stage initial hidden states into LDS as f16.
  for (int i = tid; i < RNN_N * RNN_H; i += 1024) {
    h0buf0[i] = (_Float16)h_init[i];
    h1buf[i]  = (_Float16)h_init[RNN_N * RNN_H + i];
  }
  __syncthreads();

  const int ng = wave & 15;
  const int mg = wave >> 4;
  const int n0 = ng * 32;
  const int m0 = mg * 32;
  const int roff = (lane >> 4) << 3;
  const int lc   = lane & 15;
  const int colN[2] = { n0 + lc, n0 + 16 + lc };
  const float b1v[2] = { bih1[colN[0]] + bhh1[colN[0]],
                         bih1[colN[1]] + bhh1[colN[1]] };

  for (int t = 0; t < RNN_L; ++t) {
    const _Float16* h0cur = (t & 1) ? h0buf1 : h0buf0;
    _Float16*       h0nxt = (t & 1) ? h0buf0 : h0buf1;

    // ---- layer 0: h0nxt = tanh(Y0[t] + h0cur @ Whh0^T) ----
    v8f c[2][2];
#pragma unroll
    for (int mt = 0; mt < 2; ++mt)
#pragma unroll
      for (int nt = 0; nt < 2; ++nt) {
        // Transposed Y0: 8 consecutive rows of one column = one b128 load.
        const v8h y = *(const v8h*)(Y0t + (size_t)colN[nt] * RNN_LN
                                    + t * RNN_N + m0 + mt * 16 + roff);
#pragma unroll
        for (int i = 0; i < 8; ++i) c[mt][nt][i] = (float)y[i];
      }

    for (int k0 = 0; k0 < RNN_H; k0 += 32) {
      const v16h a0 = load_frag_h16(h0cur + m0 * RNN_H + k0, RNN_H, lane);
      const v16h a1 = load_frag_h16(h0cur + (m0 + 16) * RNN_H + k0, RNN_H, lane);
      const v16h b0 = load_frag_h16(Whh0 + n0 * RNN_H + k0, RNN_H, lane);
      const v16h b1 = load_frag_h16(Whh0 + (n0 + 16) * RNN_H + k0, RNN_H, lane);
      c[0][0] = wmma_f16(a0, b0, c[0][0]);
      c[0][1] = wmma_f16(a0, b1, c[0][1]);
      c[1][0] = wmma_f16(a1, b0, c[1][0]);
      c[1][1] = wmma_f16(a1, b1, c[1][1]);
    }
#pragma unroll
    for (int mt = 0; mt < 2; ++mt)
#pragma unroll
      for (int nt = 0; nt < 2; ++nt)
#pragma unroll
        for (int i = 0; i < 8; ++i) {
          const int row = m0 + mt * 16 + roff + i;
          const float v = tanhf(c[mt][nt][i]);
          h0nxt[row * RNN_H + colN[nt]] = (_Float16)v;
          if (t == RNN_L - 1) hn[(size_t)row * RNN_H + colN[nt]] = v;
        }
    __syncthreads();  // h0nxt complete & visible to all waves

    // Prefetch next step's Y0 slab (64KB across 1024 threads) during layer-1.
    if (t + 1 < RNN_L) {
      const int pcol = tid >> 1;                 // 512 columns
      __builtin_prefetch(Y0t + (size_t)pcol * RNN_LN + (t + 1) * RNN_N
                             + (tid & 1) * 32, 0, 1);
    }

    // ---- layer 1: h1 = tanh(h0nxt @ Wih1^T + h1 @ Whh1^T + b1) ----
#pragma unroll
    for (int mt = 0; mt < 2; ++mt)
#pragma unroll
      for (int nt = 0; nt < 2; ++nt)
#pragma unroll
        for (int i = 0; i < 8; ++i) c[mt][nt][i] = b1v[nt];

    for (int k0 = 0; k0 < RNN_H; k0 += 32) {
      const v16h a0 = load_frag_h16(h0nxt + m0 * RNN_H + k0, RNN_H, lane);
      const v16h a1 = load_frag_h16(h0nxt + (m0 + 16) * RNN_H + k0, RNN_H, lane);
      const v16h b0 = load_frag_h16(Wih1 + n0 * RNN_H + k0, RNN_H, lane);
      const v16h b1 = load_frag_h16(Wih1 + (n0 + 16) * RNN_H + k0, RNN_H, lane);
      c[0][0] = wmma_f16(a0, b0, c[0][0]);
      c[0][1] = wmma_f16(a0, b1, c[0][1]);
      c[1][0] = wmma_f16(a1, b0, c[1][0]);
      c[1][1] = wmma_f16(a1, b1, c[1][1]);
      const v16h p0 = load_frag_h16(h1buf + m0 * RNN_H + k0, RNN_H, lane);
      const v16h p1 = load_frag_h16(h1buf + (m0 + 16) * RNN_H + k0, RNN_H, lane);
      const v16h q0 = load_frag_h16(Whh1 + n0 * RNN_H + k0, RNN_H, lane);
      const v16h q1 = load_frag_h16(Whh1 + (n0 + 16) * RNN_H + k0, RNN_H, lane);
      c[0][0] = wmma_f16(p0, q0, c[0][0]);
      c[0][1] = wmma_f16(p0, q1, c[0][1]);
      c[1][0] = wmma_f16(p1, q0, c[1][0]);
      c[1][1] = wmma_f16(p1, q1, c[1][1]);
    }
    // tanh, write global output now; defer LDS h1 update past the barrier.
#pragma unroll
    for (int mt = 0; mt < 2; ++mt)
#pragma unroll
      for (int nt = 0; nt < 2; ++nt)
#pragma unroll
        for (int i = 0; i < 8; ++i) {
          const int row = m0 + mt * 16 + roff + i;
          const float v = tanhf(c[mt][nt][i]);
          c[mt][nt][i] = v;  // keep tanh'd value for the LDS update
          out[((size_t)t * RNN_N + row) * RNN_H + colN[nt]] = v;
        }
    __syncthreads();  // all reads of h1buf (t-1 state) complete
#pragma unroll
    for (int mt = 0; mt < 2; ++mt)
#pragma unroll
      for (int nt = 0; nt < 2; ++nt)
#pragma unroll
        for (int i = 0; i < 8; ++i) {
          const int row = m0 + mt * 16 + roff + i;
          const float v = c[mt][nt][i];
          h1buf[row * RNN_H + colN[nt]] = (_Float16)v;
          if (t == RNN_L - 1)
            hn[(size_t)RNN_N * RNN_H + (size_t)row * RNN_H + colN[nt]] = v;
        }
    __syncthreads();  // h1buf updated before next step
  }
}

// ---------------------------------------------------------------------------
extern "C" void kernel_launch(void* const* d_in, const int* in_sizes, int n_in,
                              void* d_out, int out_size, void* d_ws, size_t ws_size,
                              hipStream_t stream) {
  (void)in_sizes; (void)n_in; (void)out_size; (void)ws_size;

  const float* input = (const float*)d_in[0];   // (L,N,I)
  const float* h_0   = (const float*)d_in[1];   // (2,N,H)
  const float* Wih0  = (const float*)d_in[2];   // (H,I)
  const float* Whh0  = (const float*)d_in[3];   // (H,H)
  const float* bih0  = (const float*)d_in[4];
  const float* bhh0  = (const float*)d_in[5];
  const float* Wih1  = (const float*)d_in[6];   // (H,H)
  const float* Whh1  = (const float*)d_in[7];   // (H,H)
  const float* bih1  = (const float*)d_in[8];
  const float* bhh1  = (const float*)d_in[9];

  float* out = (float*)d_out;                          // (L,N,H)
  float* hn  = out + (size_t)RNN_L * RNN_N * RNN_H;    // (2,N,H)

  // Workspace: Y0t f16 (32MB) | x f16 (32MB) | 4 weight matrices f16 (2MB)
  char* ws = (char*)d_ws;
  _Float16* Y0t   = (_Float16*)ws;
  _Float16* xh    = Y0t   + (size_t)RNN_LN * RNN_H;
  _Float16* Wih0h = xh    + (size_t)RNN_LN * RNN_I;
  _Float16* Whh0h = Wih0h + (size_t)RNN_H * RNN_I;
  _Float16* Wih1h = Whh0h + (size_t)RNN_H * RNN_H;
  _Float16* Whh1h = Wih1h + (size_t)RNN_H * RNN_H;

  // Phase 0: f32 -> f16 conversions (8 elements/thread).
  const int nx8 = RNN_LN * RNN_I / 8;          // 2,097,152
  const int nw8 = RNN_H * RNN_H / 8;           // 32,768
  f32_to_f16_vec8_kernel<<<nx8 / 256, 256, 0, stream>>>(input, xh, nx8);
  f32_to_f16_vec8_kernel<<<nw8 / 256, 256, 0, stream>>>(Wih0, Wih0h, nw8);
  f32_to_f16_vec8_kernel<<<nw8 / 256, 256, 0, stream>>>(Whh0, Whh0h, nw8);
  f32_to_f16_vec8_kernel<<<nw8 / 256, 256, 0, stream>>>(Wih1, Wih1h, nw8);
  f32_to_f16_vec8_kernel<<<nw8 / 256, 256, 0, stream>>>(Whh1, Whh1h, nw8);

  // Phase 1: 1024 M-groups x 8 N-groups = 8192 waves = 256 blocks.
  rnn_pre_kernel<<<256, 1024, 0, stream>>>(xh, Wih0h, bih0, bhh0, Y0t);

  // Phase 2: single persistent workgroup, 192KB dynamic LDS.
  const size_t smem_bytes = (size_t)3 * RNN_N * RNN_H * sizeof(_Float16);
  hipFuncSetAttribute((const void*)rnn_recur_kernel,
                      hipFuncAttributeMaxDynamicSharedMemorySize,
                      (int)smem_bytes);
  rnn_recur_kernel<<<1, 1024, smem_bytes, stream>>>(
      Y0t, Whh0h, Wih1h, Whh1h, bih1, bhh1, h_0, out, hn);
}
